// QuantumFeedForward_65481071410281
// MI455X (gfx1250) — matrix operand order repaired and checked by reading, not compiled, not run on hardware
//
#include <hip/hip_runtime.h>
#include <math.h>

#define EMBED  1024
#define FFN    4096
#define NQ     8
#define TOKENS 16384

#define BM 128   // rows (tokens) per workgroup
#define BN 512   // cols (embed) per workgroup
#define BK 32    // k-step (one wmma K)
#define LDH 40   // padded LDS row stride in halves (20 dwords -> conflict-free)
#define NT 512   // threads per workgroup (16 waves, 2m x 8n)

typedef __attribute__((ext_vector_type(16))) _Float16 v16h;
typedef __attribute__((ext_vector_type(8)))  _Float16 v8h;
typedef __attribute__((ext_vector_type(4)))  _Float16 v4h;
typedef __attribute__((ext_vector_type(8)))  float    v8f;

// ---------------- W2 f32 -> f16 conversion (one-time, 16MB -> 8MB) ----------
__global__ void __launch_bounds__(256) cvt_w2_kernel(const float* __restrict__ src,
                                                     _Float16* __restrict__ dst) {
    int i = blockIdx.x * blockDim.x + threadIdx.x;  // 1M threads, 4 elems each
    float4 v = ((const float4*)src)[i];
    v4h h;
    h[0] = (_Float16)v.x; h[1] = (_Float16)v.y;
    h[2] = (_Float16)v.z; h[3] = (_Float16)v.w;
    ((v4h*)dst)[i] = h;
}

// ---------------- fused  cos -> FFN1+relu -> FFN2(+bias)  -------------------
__global__ void __launch_bounds__(NT) ffn_fused_kernel(
    const float*    __restrict__ x,     // [16384, 1024]
    const float*    __restrict__ W1,    // [4096, 8]
    const float*    __restrict__ b1,    // [4096]
    const _Float16* __restrict__ W2h,   // [1024, 4096] fp16
    const float*    __restrict__ b2,    // [1024]
    float*          __restrict__ out)   // [16384, 1024]
{
    __shared__ float    qs[BM][NQ];        // cos(x[:, :8]) for this row tile
    __shared__ _Float16 hs[2][BM][LDH];    // double-buffered h chunk, fp16

    const int tid  = threadIdx.x;
    const int lane = tid & 31;
    const int wave = tid >> 5;              // 0..15
    const int m_base = blockIdx.x * BM;
    const int n_base = blockIdx.y * BN;
    const int wm = (wave >> 3) * 64;        // wave row offset (2 wave-rows)
    const int wn = (wave & 7)  * 64;        // wave col offset (8 wave-cols)

    // ---- stage q = cos(x[row, 0:8]) once per tile
    for (int i = tid; i < BM * NQ; i += NT) {
        int r = i >> 3, c = i & 7;
        qs[r][c] = cosf(x[(size_t)(m_base + r) * EMBED + c]);
    }

    v8f acc[4][4];
    const v8f vzero = {0.f, 0.f, 0.f, 0.f, 0.f, 0.f, 0.f, 0.f};
    #pragma unroll
    for (int mi = 0; mi < 4; ++mi)
        #pragma unroll
        for (int ni = 0; ni < 4; ++ni)
            acc[mi][ni] = vzero;

    // h-compute assignment: fixed k-column per thread, 8 rows each
    const int hk  = tid & 31;   // k within chunk
    const int hr0 = tid >> 5;   // rows hr0 + 16*i  (i = 0..7)

    const int laneN = lane & 15;
    const int aK    = (lane & 16) ? 8  : 0;   // A frag: lanes16-31 hold K 8..15,24..31
    const int bK    = (lane & 16) ? 8  : 0;   // B frag: lanes16-31 hold K 8..15,24..31

    // B operand base: column (n) per lane, K contiguous along W2 row
    const _Float16* Bp = W2h + (size_t)(n_base + wn + laneN) * FFN + bK;

    // h[:, kc:kc+32] = relu(q.W1^T + b1) -> dst (fp16 in LDS)
    auto compute_h = [&](int kc, _Float16 (*dst)[LDH]) {
        const float* w1r = W1 + (size_t)(kc + hk) * NQ;
        float4 wa = *(const float4*)(w1r);
        float4 wb = *(const float4*)(w1r + 4);
        float bias = b1[kc + hk];
        #pragma unroll
        for (int i = 0; i < 8; ++i) {
            int r = hr0 + (i << 4);
            float s = bias;
            s += qs[r][0]*wa.x + qs[r][1]*wa.y + qs[r][2]*wa.z + qs[r][3]*wa.w;
            s += qs[r][4]*wb.x + qs[r][5]*wb.y + qs[r][6]*wb.z + qs[r][7]*wb.w;
            dst[r][hk] = (_Float16)fmaxf(s, 0.0f);
        }
    };

    // one k-stage: consume h from `cur`, produce chunk kc+BK into `nxt`
    auto stage = [&](int kc, _Float16 (*cur)[LDH], _Float16 (*nxt)[LDH],
                     bool doNext) {
        // A fragments from LDS (ISA 16-bit A 16x32 layout)
        v16h afr[4];
        #pragma unroll
        for (int mi = 0; mi < 4; ++mi) {
            const _Float16* p = &cur[wm + mi * 16 + laneN][aK];
            ((v8h*)&afr[mi])[0] = *(const v8h*)(p);        // K 0..7   (or 8..15)
            ((v8h*)&afr[mi])[1] = *(const v8h*)(p + 16);   // K 16..23 (or 24..31)
        }

        // overlap: recompute next h chunk (VALU co-executes with WMMA below)
        if (doNext) compute_h(kc + BK, nxt);

        // B fragments from global (L2/WGP$-resident), 16 WMMAs
        #pragma unroll
        for (int ni = 0; ni < 4; ++ni) {
            v16h bfr;
            const _Float16* p = Bp + (size_t)ni * 16 * FFN + kc;
            ((v8h*)&bfr)[0] = *(const v8h*)(p);            // K 0..7   (or 8..15)
            ((v8h*)&bfr)[1] = *(const v8h*)(p + 16);       // K 16..23 (or 24..31)
            #pragma unroll
            for (int mi = 0; mi < 4; ++mi) {
                acc[mi][ni] = __builtin_amdgcn_wmma_f32_16x16x32_f16(
                    false, afr[mi], false, bfr, (short)0, acc[mi][ni], false, false);
            }
        }
        __syncthreads();   // publish nxt, retire cur (one barrier per stage)
    };

    __syncthreads();               // qs ready
    compute_h(0, hs[0]);           // prologue: h chunk 0
    __syncthreads();

    // FFN/BK = 128 stages, even -> ping/pong with compile-time buffer indices
    for (int k0 = 0; k0 < FFN; k0 += 2 * BK) {
        stage(k0,      hs[0], hs[1], true);
        stage(k0 + BK, hs[1], hs[0], k0 + 2 * BK < FFN);
    }

    // ---- epilogue: out = acc + b2   (C/D layout: VGPR r -> M=r / r+8)
    const int rOff = (lane & 16) ? 8 : 0;
    #pragma unroll
    for (int ni = 0; ni < 4; ++ni) {
        int col = n_base + wn + ni * 16 + laneN;
        float bias = b2[col];
        #pragma unroll
        for (int mi = 0; mi < 4; ++mi) {
            int row0 = m_base + wm + mi * 16 + rOff;
            #pragma unroll
            for (int r = 0; r < 8; ++r) {
                out[(size_t)(row0 + r) * EMBED + col] = acc[mi][ni][r] + bias;
            }
        }
    }
}

extern "C" void kernel_launch(void* const* d_in, const int* in_sizes, int n_in,
                              void* d_out, int out_size, void* d_ws, size_t ws_size,
                              hipStream_t stream) {
    const float* x  = (const float*)d_in[0];
    const float* W1 = (const float*)d_in[1];
    const float* b1 = (const float*)d_in[2];
    const float* W2 = (const float*)d_in[3];
    const float* b2 = (const float*)d_in[4];
    float* out = (float*)d_out;

    _Float16* W2h = (_Float16*)d_ws;   // 1024*4096 fp16 = 8 MB scratch

    // one-time weight conversion (deterministic, re-run every call)
    cvt_w2_kernel<<<(EMBED * FFN / 4) / 256, 256, 0, stream>>>(W2, W2h);

    dim3 grid(TOKENS / BM, EMBED / BN);   // 128 x 2 workgroups, 16 waves each
    ffn_fused_kernel<<<grid, NT, 0, stream>>>(x, W1, b1, W2h, b2, out);
}
